// EmbeddingModel_25159918420487
// MI455X (gfx1250) — matrix operand order repaired and checked by reading, not compiled, not run on hardware
//
#include <hip/hip_runtime.h>
#include <math.h>

// ---------------- problem constants (from reference) ----------------
#define VOCAB     100000
#define DIM       128
#define BATCH     16384
#define C_POS     10
#define NEG_N     50
#define NCTX      60          // C_POS + NEG_N
#define P_PAIRS   2000
#define LE_LAMBDA 0.01f

typedef __attribute__((ext_vector_type(16))) _Float16 v16h;
typedef __attribute__((ext_vector_type(8)))  float    v8f;

// log_sigmoid(x) = min(x,0) - log1p(exp(-|x|))   (numerically stable)
__device__ __forceinline__ float log_sigmoid(float x) {
    return fminf(x, 0.0f) - log1pf(__expf(-fabsf(x)));
}

// convert 8 consecutive f32 into halves [base..base+7] of a v16h
__device__ __forceinline__ void cvt8(v16h& h, int base, const float* __restrict__ p) {
    const float4 a = *(const float4*)(p);
    const float4 b = *(const float4*)(p + 4);
    h[base + 0] = (_Float16)a.x;  h[base + 1] = (_Float16)a.y;
    h[base + 2] = (_Float16)a.z;  h[base + 3] = (_Float16)a.w;
    h[base + 4] = (_Float16)b.x;  h[base + 5] = (_Float16)b.y;
    h[base + 6] = (_Float16)b.z;  h[base + 7] = (_Float16)b.w;
}

// ---------------------------------------------------------------------------
// Skip-gram negative-sampling loss. One wave (32 lanes) per batch element.
// WMMA trick: A[m,k] = ie[k] for all m (row-broadcast), B[:,n] = ctx row n.
// Then D[m,n] = dot(ie, ctx_n) for every m; lane n (n<16) reads c[0].
//   16-bit A 16x32 layout: lane<16 elems 0..7 = K kb+0..7, 8..15 = K kb+16..23
//                          lane>=16 shifts both K groups by +8
//   16-bit B 32x16 layout: lane n   = K kb+0..7, kb+16..23 of column n
//                          lane n+16 = K kb+8..15, kb+24..31 of column n
// ---------------------------------------------------------------------------
__global__ void __launch_bounds__(256)
sgns_loss_kernel(const int*   __restrict__ input_labels,
                 const int*   __restrict__ pos_labels,
                 const int*   __restrict__ neg_labels,
                 const float* __restrict__ in_embed,
                 const float* __restrict__ out_embed,
                 float*       __restrict__ acc)   // acc[1] += -(log_pos+log_neg) sums
{
    const int wave = (int)((blockIdx.x * blockDim.x + threadIdx.x) >> 5);
    const int lane = (int)(threadIdx.x & 31);
    if (wave >= BATCH) return;                 // wave-uniform (never taken: exact grid)

    const int b   = wave;
    const int hi  = lane >> 4;                 // 0: low half, 1: high half
    const int col = lane & 15;                 // context column handled by this lane

    const float* ie_row = in_embed + (size_t)input_labels[b] * DIM;

    // Preload A operand (ie broadcast over M) for all 4 K-steps.
    v16h ah[4];
    #pragma unroll
    for (int ks = 0; ks < 4; ++ks) {
        const int kb = ks * 32;
        ah[ks] = v16h{};
        cvt8(ah[ks], 0, ie_row + kb + 8 * hi);
        cvt8(ah[ks], 8, ie_row + kb + 16 + 8 * hi);
    }

    float lsum = 0.0f;

    #pragma unroll
    for (int t = 0; t < 4; ++t) {              // 4 tiles of 16 contexts -> 64 (60 valid)
        const int j  = t * 16 + col;
        const int jc = (j < NCTX) ? j : (NCTX - 1);       // clamp padded columns
        const int label = (jc < C_POS) ? pos_labels[b * C_POS + jc]
                                       : neg_labels[b * NEG_N + (jc - C_POS)];
        const float* ctx_row = out_embed + (size_t)label * DIM;

        v8f c = {};
        #pragma unroll
        for (int ks = 0; ks < 4; ++ks) {
            const int kb = ks * 32 + 8 * hi;
            v16h bh = v16h{};
            cvt8(bh, 0, ctx_row + kb);
            cvt8(bh, 8, ctx_row + kb + 16);
            // v_wmma_f32_16x16x32_f16 : D = A*B + C (f32 accumulate)
            c = __builtin_amdgcn_wmma_f32_16x16x32_f16(
                    /*neg_a=*/false, ah[ks], /*neg_b=*/false, bh,
                    /*c_mod=*/(short)0, c, /*reuse_a=*/false, /*reuse_b=*/false);
        }

        // D is M-redundant; lane<16: c[0] = D[0, lane] = dot(ie, ctx_j)
        if (hi == 0 && j < NCTX) {
            const float dot = c[0];
            const float x   = (j < C_POS) ? dot : -dot;   // negatives: dot with -ie
            lsum += log_sigmoid(x);
        }
    }

    // wave32 reduction; accumulate -(log_pos + log_neg)
    #pragma unroll
    for (int off = 16; off > 0; off >>= 1)
        lsum += __shfl_xor(lsum, off, 32);
    if (lane == 0) atomicAdd(acc + 1, -lsum);
}

// ---------------------------------------------------------------------------
// Hierarchy regularizer: one wave per pair, 4 dims per lane.
// ---------------------------------------------------------------------------
__global__ void __launch_bounds__(256)
pair_norm_kernel(const int*   __restrict__ pairs,
                 const float* __restrict__ in_embed,
                 float*       __restrict__ acc)   // acc[0] += ||e0 - e1||
{
    const int wave = (int)((blockIdx.x * blockDim.x + threadIdx.x) >> 5);
    const int lane = (int)(threadIdx.x & 31);
    if (wave >= P_PAIRS) return;

    const int i0 = pairs[wave * 2 + 0];
    const int i1 = pairs[wave * 2 + 1];
    const float4 a = *(const float4*)(in_embed + (size_t)i0 * DIM + lane * 4);
    const float4 b = *(const float4*)(in_embed + (size_t)i1 * DIM + lane * 4);
    const float dx = a.x - b.x, dy = a.y - b.y, dz = a.z - b.z, dw = a.w - b.w;
    float s = dx * dx + dy * dy + dz * dz + dw * dw;

    #pragma unroll
    for (int off = 16; off > 0; off >>= 1)
        s += __shfl_xor(s, off, 32);
    if (lane == 0) atomicAdd(acc + 0, sqrtf(s));
}

__global__ void init_kernel(float* acc) {
    if (threadIdx.x < 2) acc[threadIdx.x] = 0.0f;
}

__global__ void finalize_kernel(const float* __restrict__ acc, float* __restrict__ out) {
    if (threadIdx.x == 0) {
        const float loss_graph = acc[1] / (float)BATCH;
        const float s          = acc[0];
        const float loss_le    = 0.5f * LE_LAMBDA * s * s;
        out[0] = loss_graph + loss_le;   // loss_combined
        out[1] = loss_le;                // loss_le
    }
}

extern "C" void kernel_launch(void* const* d_in, const int* in_sizes, int n_in,
                              void* d_out, int out_size, void* d_ws, size_t ws_size,
                              hipStream_t stream) {
    (void)in_sizes; (void)n_in; (void)out_size; (void)ws_size;
    const int*   input_labels = (const int*)  d_in[0];
    const int*   pos_labels   = (const int*)  d_in[1];
    const int*   neg_labels   = (const int*)  d_in[2];
    const int*   pairs        = (const int*)  d_in[3];
    const float* in_embed     = (const float*)d_in[4];
    const float* out_embed    = (const float*)d_in[5];
    float* acc = (float*)d_ws;
    float* out = (float*)d_out;

    hipLaunchKernelGGL(init_kernel, dim3(1), dim3(32), 0, stream, acc);
    // 8 waves / 256-thread block, one wave per batch element: 16384/8 = 2048 blocks
    hipLaunchKernelGGL(sgns_loss_kernel, dim3(BATCH / 8), dim3(256), 0, stream,
                       input_labels, pos_labels, neg_labels, in_embed, out_embed, acc);
    // one wave per pair: 2000/8 = 250 blocks
    hipLaunchKernelGGL(pair_norm_kernel, dim3(P_PAIRS / 8), dim3(256), 0, stream,
                       pairs, in_embed, acc);
    hipLaunchKernelGGL(finalize_kernel, dim3(1), dim3(32), 0, stream, acc, out);
}
